// CustomTransformerEncoderLayer_7000796692699
// MI455X (gfx1250) — matrix operand order repaired and checked by reading, not compiled, not run on hardware
//
#include <hip/hip_runtime.h>

typedef __bf16 bf16_t;
typedef __attribute__((ext_vector_type(16))) __bf16 v16bf;
typedef __attribute__((ext_vector_type(8)))  float  v8f;
typedef unsigned int u32x4 __attribute__((ext_vector_type(4)));
typedef int          i32x4 __attribute__((ext_vector_type(4)));
typedef int          i32x8 __attribute__((ext_vector_type(8)));

#define S_DIM 2048
#define B_DIM 8
#define D_DIM 1024
#define F_DIM 4096
#define N_ROW (S_DIM * B_DIM)   // 16384 rows

#define KC        1024          // K-chunk staged in LDS per panel
#define LDS_ROW   2080          // 2048B row + 16B pad per 1KB (bank stride 8)
#define SMEM_BYTES (64 * LDS_ROW)

#if defined(__has_builtin)
#  if __has_builtin(__builtin_amdgcn_tensor_load_to_lds)
#    define HAVE_TDM 1
#  endif
#endif

enum { ACT_NONE = 0, ACT_ELU1 = 1, ACT_RELU = 2 };
enum { ST_BF16 = 0, ST_F32 = 1, ST_BF16_TRANS = 2 };

union FragU {
  v16bf v;
  uint4 q[2];
};

// ---------------------------------------------------------------- fp32 -> bf16
__global__ __launch_bounds__(256)
void cvt_f32_bf16_kernel(const float* __restrict__ in, bf16_t* __restrict__ out, int n)
{
  int i = (blockIdx.x * 256 + threadIdx.x) * 4;
  if (i < n) {
    float4 f = *reinterpret_cast<const float4*>(in + i);
    union { bf16_t h[4]; uint2 u; } p;
    p.h[0] = (bf16_t)f.x; p.h[1] = (bf16_t)f.y;
    p.h[2] = (bf16_t)f.z; p.h[3] = (bf16_t)f.w;
    *reinterpret_cast<uint2*>(out + i) = p.u;
  }
}

#if HAVE_TDM
// TDM 2D panel load: 64 rows x KC bf16 elements, global(row-major, ld_elems)
// -> LDS at offset 0 with 16B pad per 1KB (D# pad_interval=7(256dw), pad_amount=3(4dw)).
// 6-arg builtin form: (g0, g1, g2, g3, g_extra, cpol).
__device__ __forceinline__ void tdm_load_panel(const void* gptr, int ld_elems)
{
  const unsigned long long ga = (unsigned long long)(size_t)gptr;
  u32x4 g0;
  g0[0] = 1u;                                           // count=1 (valid user D#)
  g0[1] = 0u;                                           // lds_addr = 0 (dynamic LDS base)
  g0[2] = (unsigned)ga;                                 // global_addr[31:0]
  g0[3] = (unsigned)((ga >> 32) & 0x01ffffffull)        // global_addr[56:32]
        | (2u << 30);                                   // type = 2 ("image")
  const unsigned td0 = KC, td1 = 64;                    // tensor dims (OOB bounds)
  const unsigned tl0 = KC, tl1 = 64;                    // tile dims
  const unsigned long long s0 = (unsigned long long)ld_elems;  // dim0 stride
  i32x8 g1;
  g1[0] = (int)((1u << 16)        // data_size = 2 bytes
              | (1u << 20)        // pad_enable
              | (7u << 22)        // pad_interval = 256 dwords (1KB)
              | (3u << 25));      // pad_amount  = 4 dwords (16B)
  g1[1] = (int)((td0 & 0xffffu) << 16);                             // tensor_dim0[15:0]
  g1[2] = (int)(((td0 >> 16) & 0xffffu) | ((td1 & 0xffffu) << 16)); // dim0 hi | dim1 lo
  g1[3] = (int)(((td1 >> 16) & 0xffffu) | ((tl0 & 0xffffu) << 16)); // dim1 hi | tile0
  g1[4] = (int)(tl1 & 0xffffu);                                     // tile1 | tile2=0
  g1[5] = (int)(unsigned)(s0 & 0xffffffffu);                        // stride0 lo
  g1[6] = (int)(unsigned)((s0 >> 32) & 0xffffu);                    // stride0 hi | stride1 lo
  g1[7] = 0;                                                        // stride1 hi
  i32x4 z4 = {0, 0, 0, 0};
  i32x8 z8 = {0, 0, 0, 0, 0, 0, 0, 0};
  __builtin_amdgcn_tensor_load_to_lds(g0, g1, z4, z4, z8, 0);
}
#endif

// ---------------------------------------------------------------- direct WMMA GEMM
// 1 wave per 64x64 tile, batched; used for the activation x activation GEMMs.
template<int ACT, int ST>
__global__ __launch_bounds__(32)
void wmma_gemm_direct_kernel(const bf16_t* __restrict__ A, long a_batch, int lda,
                             const bf16_t* __restrict__ Bw, long b_batch, int ldb,
                             const float* __restrict__ bias,
                             void* __restrict__ Cout, long c_batch, int ldc,
                             int kdim, int sdim)
{
  const int lane = threadIdx.x;
  const int half = lane >> 4;
  const int mn   = lane & 15;
  const int row0 = blockIdx.x * 64;
  const int col0 = blockIdx.y * 64;
  const long zb  = blockIdx.z;

  const bf16_t* Ab = A + zb * a_batch;
  const bf16_t* Bb = Bw + zb * b_batch;
  const long coff  = zb * c_batch;

  v8f acc[4][4];
  #pragma unroll
  for (int i = 0; i < 4; i++)
    #pragma unroll
    for (int j = 0; j < 4; j++)
      #pragma unroll
      for (int e = 0; e < 8; e++)
        acc[i][j][e] = 0.0f;

  const bf16_t* aptr[4];
  const bf16_t* bptr[4];
  #pragma unroll
  for (int i = 0; i < 4; i++)
    aptr[i] = Ab + (long)(row0 + i * 16 + mn) * lda + 16 * half;
  #pragma unroll
  for (int j = 0; j < 4; j++)
    bptr[j] = Bb + (long)(col0 + j * 16 + mn) * ldb + 16 * half;

  for (int k0 = 0; k0 < kdim; k0 += 32) {
    FragU af[4], bg[4];
    #pragma unroll
    for (int i = 0; i < 4; i++) {
      const bf16_t* p = aptr[i] + k0;
      af[i].q[0] = *reinterpret_cast<const uint4*>(p);
      af[i].q[1] = *reinterpret_cast<const uint4*>(p + 8);
    }
    #pragma unroll
    for (int j = 0; j < 4; j++) {
      const bf16_t* p = bptr[j] + k0;
      bg[j].q[0] = *reinterpret_cast<const uint4*>(p);
      bg[j].q[1] = *reinterpret_cast<const uint4*>(p + 8);
    }
    if (k0 + 32 < kdim) {
      __builtin_prefetch(aptr[0] + k0 + 32, 0, 1);
      __builtin_prefetch(bptr[0] + k0 + 32, 0, 1);
    }
    #pragma unroll
    for (int i = 0; i < 4; i++)
      #pragma unroll
      for (int j = 0; j < 4; j++)
        acc[i][j] = __builtin_amdgcn_wmma_f32_16x16x32_bf16(
            false, af[i].v, false, bg[j].v, (short)0, acc[i][j], false, false);
  }

  #pragma unroll
  for (int i = 0; i < 4; i++) {
    #pragma unroll
    for (int j = 0; j < 4; j++) {
      const int nl = col0 + j * 16 + mn;
      const float bv = bias ? bias[nl] : 0.0f;
      #pragma unroll
      for (int e = 0; e < 8; e++) {
        const int r = row0 + i * 16 + half * 8 + e;
        float val = acc[i][j][e] + bv;
        if (ACT == ACT_ELU1) val = (val > 0.0f) ? (val + 1.0f) : __expf(val);
        if (ACT == ACT_RELU) val = fmaxf(val, 0.0f);
        if (ST == ST_F32) {
          reinterpret_cast<float*>(Cout)[coff + (long)r * ldc + nl] = val;
        } else if (ST == ST_BF16) {
          reinterpret_cast<bf16_t*>(Cout)[coff + (long)r * ldc + nl] = (bf16_t)val;
        } else {
          const long idx = ((long)(r & (B_DIM - 1)) * D_DIM + nl) * (long)sdim + (r >> 3);
          reinterpret_cast<bf16_t*>(Cout)[idx] = (bf16_t)val;
        }
      }
    }
  }
}

// ---------------------------------------------------------------- panel WMMA GEMM
// 4 waves / block; block tile = 256 rows x 64 cols. The shared 64 x KC weight
// panel is staged into LDS by the Tensor Data Mover (wave 0 issues
// tensor_load_to_lds, waits on TENSORcnt, workgroup barrier), then all waves
// read B fragments from LDS (ds_load_b128, bank-padded) while streaming A
// fragments from global/L2.
template<int ACT, int ST>
__global__ __launch_bounds__(128)
void wmma_gemm_panel_kernel(const bf16_t* __restrict__ A, int lda,
                            const bf16_t* __restrict__ Bw, int ldb,
                            const float* __restrict__ bias,
                            void* __restrict__ Cout, int ldc,
                            int kdim, int sdim)
{
  extern __shared__ __align__(16) char sm[];
  const int lane = threadIdx.x & 31;
  const int wv   = threadIdx.x >> 5;
  const int half = lane >> 4;
  const int mn   = lane & 15;
  const int row0 = blockIdx.x * 256 + wv * 64;
  const int col0 = blockIdx.y * 64;

  v8f acc[4][4];
  #pragma unroll
  for (int i = 0; i < 4; i++)
    #pragma unroll
    for (int j = 0; j < 4; j++)
      #pragma unroll
      for (int e = 0; e < 8; e++)
        acc[i][j][e] = 0.0f;

  const bf16_t* aptr[4];
  #pragma unroll
  for (int i = 0; i < 4; i++)
    aptr[i] = A + (long)(row0 + i * 16 + mn) * lda + 16 * half;

  for (int kc = 0; kc < kdim; kc += KC) {
#if HAVE_TDM
    if (threadIdx.x < 32) {                       // wave 0 drives the TDM
      tdm_load_panel(Bw + (long)col0 * ldb + kc, ldb);
      __builtin_amdgcn_s_wait_tensorcnt(0);
    }
#else
    {                                             // cooperative fallback copy
      const int row = threadIdx.x >> 1;           // 0..63
      const int hh  = threadIdx.x & 1;            // which 512-element half
      const bf16_t* gsrc = Bw + (long)(col0 + row) * ldb + kc + hh * 512;
      char* dst = sm + row * LDS_ROW + hh * (1024 + 16);
      #pragma unroll
      for (int u = 0; u < 64; u++)
        *reinterpret_cast<uint4*>(dst + u * 16) =
            *reinterpret_cast<const uint4*>(gsrc + u * 8);
    }
#endif
    __syncthreads();

    for (int k0 = 0; k0 < KC; k0 += 32) {
      FragU af[4], bg[4];
      #pragma unroll
      for (int i = 0; i < 4; i++) {
        const bf16_t* p = aptr[i] + kc + k0;
        af[i].q[0] = *reinterpret_cast<const uint4*>(p);
        af[i].q[1] = *reinterpret_cast<const uint4*>(p + 8);
      }
      #pragma unroll
      for (int j = 0; j < 4; j++) {
        const int cl  = j * 16 + mn;              // panel-local column
        const int kk0 = k0 + 16 * half;           // fragment K base (elements)
        const int kk1 = kk0 + 8;
        const unsigned o0 = cl * LDS_ROW + kk0 * 2 + 16 * (kk0 >> 9);
        const unsigned o1 = cl * LDS_ROW + kk1 * 2 + 16 * (kk1 >> 9);
        bg[j].q[0] = *reinterpret_cast<const uint4*>(sm + o0);
        bg[j].q[1] = *reinterpret_cast<const uint4*>(sm + o1);
      }
      if (k0 + 32 < KC)
        __builtin_prefetch(aptr[0] + kc + k0 + 32, 0, 1);
      #pragma unroll
      for (int i = 0; i < 4; i++)
        #pragma unroll
        for (int j = 0; j < 4; j++)
          acc[i][j] = __builtin_amdgcn_wmma_f32_16x16x32_bf16(
              false, af[i].v, false, bg[j].v, (short)0, acc[i][j], false, false);
    }
    __syncthreads();
  }

  #pragma unroll
  for (int i = 0; i < 4; i++) {
    #pragma unroll
    for (int j = 0; j < 4; j++) {
      const int nl = col0 + j * 16 + mn;
      const float bv = bias ? bias[nl] : 0.0f;
      #pragma unroll
      for (int e = 0; e < 8; e++) {
        const int r = row0 + i * 16 + half * 8 + e;
        float val = acc[i][j][e] + bv;
        if (ACT == ACT_ELU1) val = (val > 0.0f) ? (val + 1.0f) : __expf(val);
        if (ACT == ACT_RELU) val = fmaxf(val, 0.0f);
        if (ST == ST_F32) {
          reinterpret_cast<float*>(Cout)[(long)r * ldc + nl] = val;
        } else if (ST == ST_BF16) {
          reinterpret_cast<bf16_t*>(Cout)[(long)r * ldc + nl] = (bf16_t)val;
        } else { // row r = s*B + b of [S,B,*] -> store [b][n][s]
          const long idx = ((long)(r & (B_DIM - 1)) * D_DIM + nl) * (long)sdim + (r >> 3);
          reinterpret_cast<bf16_t*>(Cout)[idx] = (bf16_t)val;
        }
      }
    }
  }
}

// ---------------------------------------------------------------- residual + LayerNorm
__global__ __launch_bounds__(256)
void ln_residual_kernel(const float* __restrict__ xa, const float* __restrict__ xb,
                        const float* __restrict__ gamma, const float* __restrict__ beta,
                        float* __restrict__ outf, bf16_t* __restrict__ outb)
{
  const long base = (long)blockIdx.x * D_DIM;
  const int t = threadIdx.x;
  float vals[4];
  float s = 0.f, s2 = 0.f;
  #pragma unroll
  for (int i = 0; i < 4; i++) {
    const int c = t + i * 256;
    const float v = xa[base + c] + xb[base + c];
    vals[i] = v; s += v; s2 += v * v;
  }
  #pragma unroll
  for (int off = 16; off > 0; off >>= 1) {   // wave32 butterfly
    s  += __shfl_xor(s, off, 32);
    s2 += __shfl_xor(s2, off, 32);
  }
  __shared__ float red[2][8];
  const int wid = t >> 5;
  if ((t & 31) == 0) { red[0][wid] = s; red[1][wid] = s2; }
  __syncthreads();
  s = 0.f; s2 = 0.f;
  #pragma unroll
  for (int w = 0; w < 8; w++) { s += red[0][w]; s2 += red[1][w]; }
  const float mu   = s * (1.0f / D_DIM);
  const float var  = s2 * (1.0f / D_DIM) - mu * mu;
  const float rstd = rsqrtf(var + 1e-5f);
  #pragma unroll
  for (int i = 0; i < 4; i++) {
    const int c = t + i * 256;
    const float v = (vals[i] - mu) * rstd * gamma[c] + beta[c];
    outf[base + c] = v;
    if (outb) outb[base + c] = (bf16_t)v;
  }
}

// ---------------------------------------------------------------- host
extern "C" void kernel_launch(void* const* d_in, const int* in_sizes, int n_in,
                              void* d_out, int out_size, void* d_ws, size_t ws_size,
                              hipStream_t stream)
{
  (void)in_sizes; (void)n_in; (void)out_size; (void)ws_size;
  const float* src = (const float*)d_in[0];
  const float* wq  = (const float*)d_in[1];
  const float* bq  = (const float*)d_in[2];
  const float* wk  = (const float*)d_in[3];
  const float* bk  = (const float*)d_in[4];
  const float* wv  = (const float*)d_in[5];
  const float* bv  = (const float*)d_in[6];
  const float* wo  = (const float*)d_in[7];
  const float* bo  = (const float*)d_in[8];
  const float* w1  = (const float*)d_in[9];
  const float* b1  = (const float*)d_in[10];
  const float* w2  = (const float*)d_in[11];
  const float* b2  = (const float*)d_in[12];
  const float* g1  = (const float*)d_in[13];
  const float* be1 = (const float*)d_in[14];
  const float* g2  = (const float*)d_in[15];
  const float* be2 = (const float*)d_in[16];

  // ---- workspace layout (360 MB total, phase-aliased) ----
  const size_t MB = 1ull << 20;
  char* ws = (char*)d_ws;
  bf16_t* Wq   = (bf16_t*)(ws +   0 * MB);  //  2 MB (persistent)
  bf16_t* Wk   = (bf16_t*)(ws +   2 * MB);  //  2 MB
  bf16_t* Wv   = (bf16_t*)(ws +   4 * MB);  //  2 MB
  bf16_t* Wo   = (bf16_t*)(ws +   6 * MB);  //  2 MB
  bf16_t* W1   = (bf16_t*)(ws +   8 * MB);  //  8 MB
  bf16_t* W2   = (bf16_t*)(ws +  16 * MB);  //  8 MB
  bf16_t* Xb   = (bf16_t*)(ws +  24 * MB);  // 32 MB  src bf16  [N,D]
  bf16_t* Qb   = (bf16_t*)(ws +  56 * MB);  // 32 MB  Q bf16    [S,B,D]
  bf16_t* Kt   = (bf16_t*)(ws +  88 * MB);  // 32 MB  K bf16    [B,D,S]
  bf16_t* Vt   = (bf16_t*)(ws + 120 * MB);  // 32 MB  V bf16    [B,D,S]
  bf16_t* KVt  = (bf16_t*)(ws + 152 * MB);  // 16 MB  KV^T bf16 [B,E,D]
  bf16_t* Attn = (bf16_t*)(ws +  24 * MB);  // alias Xb (dead)  [S,B,E]
  float*  AOf  = (float*) (ws +  56 * MB);  // 64 MB, alias Qb+Kt (dead)
  float*  X1f  = (float*) (ws + 168 * MB);  // 64 MB  LN1 out f32
  bf16_t* X1b  = (bf16_t*)(ws + 120 * MB);  // alias Vt (dead)  LN1 out bf16
  bf16_t* Hb   = (bf16_t*)(ws + 232 * MB);  // 128 MB FFN hidden bf16 [N,F]
  float*  F2f  = (float*) (ws +  24 * MB);  // alias Attn+AOf (dead), 64 MB

  auto cvt = [&](const float* in, bf16_t* out, long n) {
    cvt_f32_bf16_kernel<<<(int)((n / 4 + 255) / 256), 256, 0, stream>>>(in, out, (int)n);
  };
  cvt(src, Xb, (long)N_ROW * D_DIM);
  cvt(wq, Wq, (long)D_DIM * D_DIM);
  cvt(wk, Wk, (long)D_DIM * D_DIM);
  cvt(wv, Wv, (long)D_DIM * D_DIM);
  cvt(wo, Wo, (long)D_DIM * D_DIM);
  cvt(w1, W1, (long)F_DIM * D_DIM);
  cvt(w2, W2, (long)D_DIM * F_DIM);

  // Q = elu(X Wq^T + bq) + 1            [N,D]
  wmma_gemm_panel_kernel<ACT_ELU1, ST_BF16><<<dim3(N_ROW/256, D_DIM/64, 1), 128, SMEM_BYTES, stream>>>(
      Xb, D_DIM, Wq, D_DIM, bq, Qb, D_DIM, D_DIM, 0);
  // K = elu(X Wk^T + bk) + 1  stored as Kt[b][d][s]
  wmma_gemm_panel_kernel<ACT_ELU1, ST_BF16_TRANS><<<dim3(N_ROW/256, D_DIM/64, 1), 128, SMEM_BYTES, stream>>>(
      Xb, D_DIM, Wk, D_DIM, bk, Kt, 0, D_DIM, S_DIM);
  // V = X Wv^T + bv           stored as Vt[b][e][s]
  wmma_gemm_panel_kernel<ACT_NONE, ST_BF16_TRANS><<<dim3(N_ROW/256, D_DIM/64, 1), 128, SMEM_BYTES, stream>>>(
      Xb, D_DIM, Wv, D_DIM, bv, Vt, 0, D_DIM, S_DIM);
  // KVt[b][e][d] = sum_s Vt[b][e][s] * Kt[b][d][s]       (direct, batched)
  wmma_gemm_direct_kernel<ACT_NONE, ST_BF16><<<dim3(D_DIM/64, D_DIM/64, B_DIM), 32, 0, stream>>>(
      Vt, (long)D_DIM * S_DIM, S_DIM, Kt, (long)D_DIM * S_DIM, S_DIM, nullptr,
      KVt, (long)D_DIM * D_DIM, D_DIM, S_DIM, 0);
  // Attn[s][b][e] = sum_d Q[s][b][d] * KVt[b][e][d]      (direct, batched)
  wmma_gemm_direct_kernel<ACT_NONE, ST_BF16><<<dim3(S_DIM/64, D_DIM/64, B_DIM), 32, 0, stream>>>(
      Qb, (long)D_DIM, B_DIM * D_DIM, KVt, (long)D_DIM * D_DIM, D_DIM, nullptr,
      Attn, (long)D_DIM, B_DIM * D_DIM, D_DIM, 0);
  // AOf = Attn Wo^T + bo  (f32)
  wmma_gemm_panel_kernel<ACT_NONE, ST_F32><<<dim3(N_ROW/256, D_DIM/64, 1), 128, SMEM_BYTES, stream>>>(
      Attn, D_DIM, Wo, D_DIM, bo, AOf, D_DIM, D_DIM, 0);
  // x1 = LN(src + attn_out)
  ln_residual_kernel<<<N_ROW, 256, 0, stream>>>(src, AOf, g1, be1, X1f, X1b);
  // H = relu(x1 W1^T + b1)              [N,F]
  wmma_gemm_panel_kernel<ACT_RELU, ST_BF16><<<dim3(N_ROW/256, F_DIM/64, 1), 128, SMEM_BYTES, stream>>>(
      X1b, D_DIM, W1, D_DIM, b1, Hb, F_DIM, D_DIM, 0);
  // F2f = H W2^T + b2 (f32)
  wmma_gemm_panel_kernel<ACT_NONE, ST_F32><<<dim3(N_ROW/256, D_DIM/64, 1), 128, SMEM_BYTES, stream>>>(
      Hb, F_DIM, W2, F_DIM, b2, F2f, D_DIM, F_DIM, 0);
  // out = LN(x1 + ffn)
  ln_residual_kernel<<<N_ROW, 256, 0, stream>>>(X1f, F2f, g2, be2, (float*)d_out, nullptr);
}
